// GCN_14671608283854
// MI455X (gfx1250) — matrix operand order repaired and checked by reading, not compiled, not run on hardware
//
#include <hip/hip_runtime.h>
#include <hip/hip_bf16.h>
#include <math.h>

// ---------------------------------------------------------------------------
// Types for CDNA5 WMMA
// ---------------------------------------------------------------------------
typedef __attribute__((ext_vector_type(16))) __bf16 v16bf;
typedef __attribute__((ext_vector_type(8)))  float  v8f;

#define DIMF   384
#define RANK   128
#define DLAT   64
#define MTILE  64
#define S1     40    // LDS elem stride for K-chunk buffers (80B = 5*16B)
#define SP     136   // LDS elem stride for p / Wproj (272B = 17*16B)
#define SZ     66    // LDS elem stride for z (f32)

__device__ __forceinline__ unsigned pack_bf16x2(float lo, float hi)
{
    union { __bf16 b[2]; unsigned u; } t;
    t.b[0] = (__bf16)lo; t.b[1] = (__bf16)hi;
    return t.u;
}

// Device-scope f32 atomic add, no return (STOREcnt-tracked). Guarantees the
// native global_atomic_add_f32 path (RMW at the device-wide L2).
__device__ __forceinline__ void atomic_fadd_dev(float* p, float v)
{
    asm volatile("global_atomic_add_f32 %0, %1, off scope:SCOPE_DEV"
                 :: "v"(p), "v"(v) : "memory");
}

// ---------------------------------------------------------------------------
// Item-side fused kernel:
//   a = F Wu^T ; b = F Wv^T ; z = (a*b) Wproj^T ; signed-sqrt ; l2norm ;
//   t = id_emb * z ; x[num_user + i] = l2norm(t)
// Block: 128 threads (4 waves), M-tile = 64 rows, bf16 WMMA 16x16x32,
// K streamed in 32-chunks with software-pipelined global->LDS staging.
// ---------------------------------------------------------------------------
__global__ __launch_bounds__(128)
void gcn_item_kernel(const float* __restrict__ feat,
                     const float* __restrict__ Wu,
                     const float* __restrict__ Wv,
                     const float* __restrict__ Wp,
                     const float* __restrict__ id_emb,
                     float* __restrict__ xbuf,
                     int num_item, int num_user)
{
    __shared__ __align__(16) char smem[34816];
    __bf16* featC = (__bf16*)(smem);           // [64][S1]   phase 1
    __bf16* wuC   = (__bf16*)(smem + 5120);    // [128][S1]  phase 1
    __bf16* wvC   = (__bf16*)(smem + 15360);   // [128][S1]  phase 1
    __bf16* pbuf  = (__bf16*)(smem);           // [64][SP]   phase 2
    __bf16* wpC   = (__bf16*)(smem + 17408);   // [64][SP]   phase 2
    float*  zbuf  = (float*)(smem);            // [64][SZ]   phase 3

    const int tid  = threadIdx.x;
    const int wave = tid >> 5;
    const int lane = tid & 31;
    const int half = lane >> 4;
    const int lm   = lane & 15;
    const int m0   = wave * 16;                  // wave's row offset in tile
    const int tileRow = blockIdx.x * MTILE;

    v8f accA[8], accB[8];
#pragma unroll
    for (int i = 0; i < 8; ++i) { accA[i] = v8f{}; accB[i] = v8f{}; }

    // staging registers (raw f32 pairs, converted after the WMMA section)
    float2 fF[8], fU[16], fV[16];

    auto issue_loads = [&](int kc) {
#pragma unroll
        for (int j = 0; j < 8; ++j) {            // features: 64x16 pairs
            int p = tid + j * 128;
            int r = p >> 4, k2 = p & 15;
            int gr = tileRow + r;
            fF[j] = (gr < num_item)
                  ? *(const float2*)(feat + (size_t)gr * DIMF + kc + k2 * 2)
                  : make_float2(0.f, 0.f);
        }
#pragma unroll
        for (int j = 0; j < 16; ++j) {           // Wu/Wv: 128x16 pairs
            int p = tid + j * 128;
            int n = p >> 4, k2 = p & 15;
            fU[j] = *(const float2*)(Wu + (size_t)n * DIMF + kc + k2 * 2);
            fV[j] = *(const float2*)(Wv + (size_t)n * DIMF + kc + k2 * 2);
        }
    };

    auto store_chunk = [&]() {
        unsigned* fS = (unsigned*)featC;
        unsigned* uS = (unsigned*)wuC;
        unsigned* vS = (unsigned*)wvC;
#pragma unroll
        for (int j = 0; j < 8; ++j) {
            int p = tid + j * 128;
            int r = p >> 4, k2 = p & 15;
            fS[r * (S1 / 2) + k2] = pack_bf16x2(fF[j].x, fF[j].y);
        }
#pragma unroll
        for (int j = 0; j < 16; ++j) {
            int p = tid + j * 128;
            int n = p >> 4, k2 = p & 15;
            uS[n * (S1 / 2) + k2] = pack_bf16x2(fU[j].x, fU[j].y);
            vS[n * (S1 / 2) + k2] = pack_bf16x2(fV[j].x, fV[j].y);
        }
    };

    // ---------------- GEMM1: a = F Wu^T, b = F Wv^T (pipelined K) ---------
    issue_loads(0);
    store_chunk();
    __syncthreads();

    for (int c = 0; c < DIMF / 32; ++c) {
        if (c + 1 < DIMF / 32) issue_loads((c + 1) * 32);  // overlap w/ WMMA

        // A fragment (16x32 bf16), contiguous k = half*16 + e (matches B)
        v16bf af;
#pragma unroll
        for (int e = 0; e < 16; ++e) {
            int k = half * 16 + e;
            af[e] = featC[(m0 + lm) * S1 + k];
        }
#pragma unroll
        for (int nt = 0; nt < 8; ++nt) {
            v16bf bu, bv;
#pragma unroll
            for (int e = 0; e < 16; ++e) {
                int k = half * 16 + e;
                bu[e] = wuC[(nt * 16 + lm) * S1 + k];
                bv[e] = wvC[(nt * 16 + lm) * S1 + k];
            }
            accA[nt] = __builtin_amdgcn_wmma_f32_16x16x32_bf16(
                false, af, false, bu, (short)0, accA[nt], false, false);
            accB[nt] = __builtin_amdgcn_wmma_f32_16x16x32_bf16(
                false, af, false, bv, (short)0, accB[nt], false, false);
        }
        __syncthreads();
        if (c + 1 < DIMF / 32) {
            store_chunk();
            __syncthreads();
        }
    }

    // ---------------- p = a*b (same C/D layout), restage as A-operand ----
#pragma unroll
    for (int nt = 0; nt < 8; ++nt) {
#pragma unroll
        for (int r = 0; r < 8; ++r) {
            float pv = accA[nt][r] * accB[nt][r];
            int rowp = m0 + half * 8 + r;
            pbuf[rowp * SP + nt * 16 + lm] = (__bf16)pv;
        }
    }
    // stage Wproj (64x128) as packed bf16x2
    {
        unsigned* wpS = (unsigned*)wpC;
#pragma unroll
        for (int j = 0; j < 32; ++j) {
            int p = tid + j * 128;                // 64x64 pairs
            int n = p >> 6, k2 = p & 63;
            float2 w = *(const float2*)(Wp + (size_t)n * RANK + k2 * 2);
            wpS[n * (SP / 2) + k2] = pack_bf16x2(w.x, w.y);
        }
    }
    __syncthreads();

    // ---------------- GEMM2: z = p Wproj^T ------------------------------
    v8f accZ[4];
#pragma unroll
    for (int i = 0; i < 4; ++i) accZ[i] = v8f{};
#pragma unroll
    for (int kc = 0; kc < RANK; kc += 32) {
        v16bf af;
#pragma unroll
        for (int e = 0; e < 16; ++e) {
            int k = kc + half * 16 + e;
            af[e] = pbuf[(m0 + lm) * SP + k];
        }
#pragma unroll
        for (int nt = 0; nt < 4; ++nt) {
            v16bf bw;
#pragma unroll
            for (int e = 0; e < 16; ++e)
                bw[e] = wpC[(nt * 16 + lm) * SP + kc + half * 16 + e];
            accZ[nt] = __builtin_amdgcn_wmma_f32_16x16x32_bf16(
                false, af, false, bw, (short)0, accZ[nt], false, false);
        }
    }
    __syncthreads();

    // dump z to LDS (f32) for the per-row epilogue
#pragma unroll
    for (int nt = 0; nt < 4; ++nt)
#pragma unroll
        for (int r = 0; r < 8; ++r)
            zbuf[(m0 + half * 8 + r) * SZ + nt * 16 + lm] = accZ[nt][r];
    __syncthreads();

    // ---------------- epilogue: signed-sqrt, l2norm, *id_emb, l2norm ----
    if (tid < MTILE) {
        int gr = tileRow + tid;
        if (gr < num_item) {
            float* zr = zbuf + tid * SZ;
            float ss = 0.f;
            for (int d = 0; d < DLAT; ++d) {
                float zv = zr[d];
                float sg = (zv > 0.f) ? 1.f : ((zv < 0.f) ? -1.f : 0.f);
                float t  = sg * sqrtf(fabsf(zv) + 1e-9f);
                zr[d] = t;
                ss += t * t;
            }
            float sc = 1.f / fmaxf(sqrtf(ss), 1e-12f);
            const float* idr = id_emb + (size_t)gr * DLAT;
            float ss2 = 0.f;
            for (int d = 0; d < DLAT; ++d) {
                float t = idr[d] * zr[d] * sc;
                zr[d] = t;
                ss2 += t * t;
            }
            float sc2 = 1.f / fmaxf(sqrtf(ss2), 1e-12f);
            float* xo = xbuf + (size_t)(num_user + gr) * DLAT;
            for (int d = 0; d < DLAT; ++d) xo[d] = zr[d] * sc2;
        }
    }
}

// ---------------------------------------------------------------------------
// Users: x[i] = l2norm(preference[i])
// ---------------------------------------------------------------------------
__global__ void gcn_user_kernel(const float* __restrict__ pref,
                                float* __restrict__ xbuf, int num_user)
{
    int r = blockIdx.x * blockDim.x + threadIdx.x;
    if (r >= num_user) return;
    const float4* p4 = (const float4*)(pref + (size_t)r * DLAT);
    float4 v[16];
    float ss = 0.f;
#pragma unroll
    for (int i = 0; i < 16; ++i) {
        v[i] = p4[i];
        ss += v[i].x * v[i].x + v[i].y * v[i].y + v[i].z * v[i].z + v[i].w * v[i].w;
    }
    float sc = 1.f / fmaxf(sqrtf(ss), 1e-12f);
    float4* o4 = (float4*)(xbuf + (size_t)r * DLAT);
#pragma unroll
    for (int i = 0; i < 16; ++i) {
        float4 o; o.x = v[i].x * sc; o.y = v[i].y * sc;
        o.z = v[i].z * sc; o.w = v[i].w * sc;
        o4[i] = o;
    }
}

// ---------------------------------------------------------------------------
// Small utility kernels
// ---------------------------------------------------------------------------
__global__ void gcn_zero_kernel(float4* __restrict__ p, int n4)
{
    int i = blockIdx.x * blockDim.x + threadIdx.x;
    if (i < n4) p[i] = make_float4(0.f, 0.f, 0.f, 0.f);
}

__global__ void gcn_zero_scalar_kernel(float* __restrict__ p, int n)
{
    int i = blockIdx.x * blockDim.x + threadIdx.x;
    if (i < n) p[i] = 0.f;
}

__global__ void gcn_deg_kernel(const int* __restrict__ row,
                               float* __restrict__ deg, int E)
{
    int e = blockIdx.x * blockDim.x + threadIdx.x;
    if (e < E) atomic_fadd_dev(&deg[row[e]], 1.0f);
}

__global__ void gcn_dinv_kernel(float* __restrict__ deg, int n)
{
    int i = blockIdx.x * blockDim.x + threadIdx.x;
    if (i < n) {
        float d = deg[i];
        deg[i] = (d > 0.f) ? rsqrtf(d) : 0.f;
    }
}

// out[col[e]] += dinv[row[e]]*dinv[col[e]] * x[row[e]]  (one edge per wave)
__global__ void gcn_scatter_kernel(const int* __restrict__ row,
                                   const int* __restrict__ col,
                                   const float* __restrict__ dinv,
                                   const float* __restrict__ x,
                                   float* __restrict__ out, int E)
{
    int t = blockIdx.x * blockDim.x + threadIdx.x;
    int e = t >> 5;
    int lane = t & 31;
    if (e >= E) return;
    int r = row[e], c = col[e];
    float nm = dinv[r] * dinv[c];
    const float2* xr = (const float2*)(x + (size_t)r * DLAT);
    float2 v = xr[lane];
    float* o = out + (size_t)c * DLAT + lane * 2;
    atomic_fadd_dev(o,     nm * v.x);
    atomic_fadd_dev(o + 1, nm * v.y);
}

__global__ void gcn_add_kernel(const float4* __restrict__ a,
                               const float4* __restrict__ b,
                               float4* __restrict__ o, int n4)
{
    int i = blockIdx.x * blockDim.x + threadIdx.x;
    if (i < n4) {
        float4 x = a[i], h = b[i], r;
        r.x = x.x + h.x; r.y = x.y + h.y; r.z = x.z + h.z; r.w = x.w + h.w;
        o[i] = r;
    }
}

// ---------------------------------------------------------------------------
// Launch sequence
// ---------------------------------------------------------------------------
extern "C" void kernel_launch(void* const* d_in, const int* in_sizes, int n_in,
                              void* d_out, int out_size, void* d_ws, size_t ws_size,
                              hipStream_t stream)
{
    const int*   edge   = (const int*)d_in[0];
    const float* feat   = (const float*)d_in[1];
    const float* Wu     = (const float*)d_in[2];
    const float* Wv     = (const float*)d_in[3];
    const float* Wp     = (const float*)d_in[4];
    const float* id_emb = (const float*)d_in[5];
    const float* pref   = (const float*)d_in[6];

    const int E        = in_sizes[0] / 2;        // 4M directed edges
    const int num_item = in_sizes[5] / DLAT;     // 200000
    const int num_user = in_sizes[6] / DLAT;     // 200000
    const int N        = num_user + num_item;    // 400000

    const int* row = edge;
    const int* col = edge + E;

    float* xbuf = (float*)d_ws;                  // [N*64]
    float* hbuf = xbuf + (size_t)N * DLAT;       // [N*64]
    float* dinv = hbuf + (size_t)N * DLAT;       // [N] (deg then dinv in-place)
    float* out  = (float*)d_out;                 // [N*64] x_hat ++ [num_user*64] pref

    const int nx = N * DLAT;

    // zero accumulators
    gcn_zero_kernel<<<(nx / 4 + 255) / 256, 256, 0, stream>>>((float4*)hbuf, nx / 4);
    gcn_zero_scalar_kernel<<<(N + 255) / 256, 256, 0, stream>>>(dinv, N);

    // x rows (items via WMMA pipeline, users via normalize)
    gcn_item_kernel<<<(num_item + MTILE - 1) / MTILE, 128, 0, stream>>>(
        feat, Wu, Wv, Wp, id_emb, xbuf, num_item, num_user);
    gcn_user_kernel<<<(num_user + 255) / 256, 256, 0, stream>>>(pref, xbuf, num_user);

    // second output: preference passthrough
    hipMemcpyAsync(out + (size_t)N * DLAT, pref,
                   (size_t)num_user * DLAT * sizeof(float),
                   hipMemcpyDeviceToDevice, stream);

    // degree-based norm
    gcn_deg_kernel<<<(E + 255) / 256, 256, 0, stream>>>(row, dinv, E);
    gcn_dinv_kernel<<<(N + 255) / 256, 256, 0, stream>>>(dinv, N);

    // layer 1: h = A x   (one edge per wave => E waves)
    {
        long long threads = (long long)E * 32;
        int blocks = (int)((threads + 255) / 256);
        gcn_scatter_kernel<<<blocks, 256, 0, stream>>>(row, col, dinv, xbuf, hbuf, E);
    }

    // out = x + h
    gcn_add_kernel<<<(nx / 4 + 255) / 256, 256, 0, stream>>>(
        (const float4*)xbuf, (const float4*)hbuf, (float4*)out, nx / 4);

    // layer 2: out += A h
    {
        long long threads = (long long)E * 32;
        int blocks = (int)((threads + 255) / 256);
        gcn_scatter_kernel<<<blocks, 256, 0, stream>>>(row, col, dinv, hbuf, out, E);
    }
}
